// CoTrackerThree_60043642798188
// MI455X (gfx1250) — matrix operand order repaired and checked by reading, not compiled, not run on hardware
//
#include <hip/hip_runtime.h>
#include <math.h>
#include <stdint.h>

typedef _Float16 half_t;
typedef __attribute__((ext_vector_type(16))) _Float16 v16h;
typedef __attribute__((ext_vector_type(8)))  float    v8f;

namespace ct {
constexpr int T = 16, H = 384, W = 512, NTRK = 128;
constexpr int LATENT = 128, HID = 384, HEADS = 8, HDIM = 48;
constexpr int ITERS = 4;
constexpr int P = 49, CORR_DIM = 196;
constexpr int Hf = 96, Wf = 128;
constexpr int NVIRT = 64;
constexpr int H1 = 192, W1 = 256, C1CH = 64;
constexpr int SEQSP = NTRK + NVIRT;  // 192
constexpr int TOKENS = NTRK * T;     // 2048
constexpr int TOKSP  = T * SEQSP;    // 3072
}
using namespace ct;

// ---------------- device helpers ----------------

static __device__ __forceinline__ float gelu_f(float x) {
  float x3 = x * x * x;
  return 0.5f * x * (1.f + tanhf(0.7978845608028654f * (x + 0.044715f * x3)));
}
static __device__ __forceinline__ int clampi(int v, int lo, int hi) {
  return v < lo ? lo : (v > hi ? hi : v);
}

// ---------------- WMMA GEMM ----------------
// A: M x Kp row-major f16 (K zero-padded to Kp, multiple of 32)
// B: Kp x Np row-major f16 (Np = N padded to multiple of 32; pad rows/cols zero)
// C: M x Np f32 (leading dim Np).  Nc = logical N (bias/store bound).
// Each wave computes a 16x32 tile = two independent 16x16 WMMA accumulator
// chains sharing one A fragment (2x arithmetic intensity, 2 chains of ILP).
// act: 0=none 1=gelu.  resid: 1 -> C += result (only used when Nc==Np).
// M must be a multiple of 16 (true for all call sites).
union V16U { v16h v; uint32_t u[8]; };

__global__ __launch_bounds__(256)
void gemm_wmma_kernel(const half_t* __restrict__ A, const half_t* __restrict__ Bm,
                      const float* __restrict__ bias, float* __restrict__ C,
                      int M, int Nc, int Np, int Kp, int act, int resid)
{
  int wave = threadIdx.x >> 5;
  int lane = threadIdx.x & 31;
  int tilesN = Np >> 5;                  // 32-wide tiles
  long tile = (long)blockIdx.x * 8 + wave;
  long nTiles = (long)(M >> 4) * tilesN;
  if (tile >= nTiles) return;            // wave-uniform: EXEC stays all-ones for WMMA
  int tm = (int)(tile / tilesN);
  int tn = (int)(tile % tilesN);
  int lo = lane & 15, hi = lane >> 4;

  // A: lane (lo,hi) -> row m=lo; dword j holds K = (j>=4?16:0)+(hi?8:0)+2*(j&3)
  const half_t* Ap = A + (long)(tm * 16 + lo) * Kp + (hi << 3);
  // B: lane (lo,hi) -> column n=lo (and n=lo+16); element s holds K = (s>=8?16:0)+(hi?8:0)+(s&7)
  const half_t* Bp = Bm + (long)(hi << 3) * Np + tn * 32 + lo;

  v8f acc0 = {}, acc1 = {};
  for (int k0 = 0; k0 < Kp; k0 += 32) {
    V16U av, bv0, bv1;
#pragma unroll
    for (int j = 0; j < 8; ++j) {
      int k = ((j & 3) << 1) + ((j >> 2) << 4);
      av.u[j] = *(const uint32_t*)(Ap + k);        // 2x global_load_b128
    }
    const half_t* br = Bp + (long)k0 * Np;
#pragma unroll
    for (int s = 0; s < 16; ++s) {
      long kk = (long)((s & 7) + ((s >> 3) << 4)) * Np;
      bv0.v[s] = br[kk];
      bv1.v[s] = br[kk + 16];
    }
    Ap += 32;
    acc0 = __builtin_amdgcn_wmma_f32_16x16x32_f16(false, av.v, false, bv0.v,
                                                  (short)0, acc0, false, false);
    acc1 = __builtin_amdgcn_wmma_f32_16x16x32_f16(false, av.v, false, bv1.v,
                                                  (short)0, acc1, false, false);
  }

  int col0 = tn * 32 + lo;
#pragma unroll
  for (int half = 0; half < 2; ++half) {
    int col = col0 + half * 16;
    const v8f& acc = half ? acc1 : acc0;
#pragma unroll
    for (int r = 0; r < 8; ++r) {
      int row = tm * 16 + r + (hi << 3);
      long idx = (long)row * Np + col;
      if (col < Nc) {
        float v = acc[r];
        if (bias) v += bias[col];
        if (act == 1) v = gelu_f(v);
        if (resid) v += C[idx];
        C[idx] = v;
      } else {
        C[idx] = 0.f;                    // keep pad columns deterministic
      }
    }
  }
}

// ---------------- casts / weight prep ----------------

// src: rows x cols f32 with leading dim srcld -> dst: rows x pcols f16, zero pad
__global__ __launch_bounds__(256)
void cast_pad_kernel(const float* __restrict__ src, half_t* __restrict__ dst,
                     long rows, int cols, int srcld, int pcols)
{
  long idx = (long)blockIdx.x * blockDim.x + threadIdx.x;
  if (idx >= rows * pcols) return;
  int c = (int)(idx % pcols);
  long r = idx / pcols;
  dst[idx] = (c < cols) ? (half_t)src[r * srcld + c] : (half_t)0.f;
}

// w: K x N f32 -> dst: Kp x Np f16 row-major, zero pad rows/cols
__global__ __launch_bounds__(256)
void castW_kernel(const float* __restrict__ w, half_t* __restrict__ dst,
                  int K, int Nc, int Kp, int Np)
{
  long idx = (long)blockIdx.x * blockDim.x + threadIdx.x;
  if (idx >= (long)Kp * Np) return;
  int n = (int)(idx % Np);
  long k = idx / Np;
  dst[idx] = (k < K && n < Nc) ? (half_t)w[k * Nc + n] : (half_t)0.f;
}

// ---------------- feature net ----------------

__global__ __launch_bounds__(256)
void conv1_kernel(const float* __restrict__ video, const float* __restrict__ w,
                  const float* __restrict__ bias, half_t* __restrict__ out)
{
  long idx = (long)blockIdx.x * blockDim.x + threadIdx.x;
  const long total = (long)T * H1 * W1 * C1CH;
  if (idx >= total) return;
  int c = (int)(idx & 63);
  long r = idx >> 6;
  int x = (int)(r % W1); r /= W1;
  int y = (int)(r % H1); r /= H1;
  int t = (int)r;
  float acc = bias[c];
  const float* vt = video + (long)t * H * W * 3;
  for (int ky = 0; ky < 7; ++ky) {
    int iy = y * 2 - 2 + ky;
    if ((unsigned)iy >= (unsigned)H) continue;
    for (int kx = 0; kx < 7; ++kx) {
      int ix = x * 2 - 2 + kx;
      if ((unsigned)ix >= (unsigned)W) continue;
      const float* px = vt + ((long)iy * W + ix) * 3;
      const float* wp = w + ((ky * 7 + kx) * 3) * 64 + c;
#pragma unroll
      for (int ci = 0; ci < 3; ++ci)
        acc += (px[ci] * 2.f - 1.f) * wp[ci * 64];
    }
  }
  out[idx] = (half_t)fmaxf(acc, 0.f);
}

__global__ __launch_bounds__(256)
void conv2_kernel(const half_t* __restrict__ in, const float* __restrict__ w,
                  const float* __restrict__ bias, half_t* __restrict__ out)
{
  long idx = (long)blockIdx.x * blockDim.x + threadIdx.x;
  const long total = (long)T * Hf * Wf * 128;
  if (idx >= total) return;
  int c = (int)(idx & 127);
  long r = idx >> 7;
  int x = (int)(r % Wf); r /= Wf;
  int y = (int)(r % Hf); r /= Hf;
  int t = (int)r;
  float acc = bias[c];
  const half_t* it = in + (long)t * H1 * W1 * 64;
  for (int ky = 0; ky < 3; ++ky) {
    int iy = y * 2 + ky;
    if (iy >= H1) continue;
    for (int kx = 0; kx < 3; ++kx) {
      int ix = x * 2 + kx;
      if (ix >= W1) continue;
      const half_t* px = it + ((long)iy * W1 + ix) * 64;
      const float* wp = w + ((ky * 3 + kx) * 64) * 128 + c;
#pragma unroll 8
      for (int ci = 0; ci < 64; ++ci)
        acc += (float)px[ci] * wp[ci * 128];
    }
  }
  out[idx] = (half_t)fmaxf(acc, 0.f);
}

__global__ __launch_bounds__(256)
void avgpool_kernel(const float* __restrict__ in, float* __restrict__ out,
                    int hin, int win, long total)
{
  long idx = (long)blockIdx.x * blockDim.x + threadIdx.x;
  if (idx >= total) return;
  int wo = win >> 1, ho = hin >> 1;
  int c = (int)(idx & 127);
  long r = idx >> 7;
  int xo = (int)(r % wo); r /= wo;
  int yo = (int)(r % ho); r /= ho;
  int t = (int)r;
  const float* base = in + (((long)t * hin + 2 * yo) * win + 2 * xo) * 128 + c;
  float s = base[0] + base[128] + base[(long)win * 128] + base[(long)win * 128 + 128];
  out[idx] = 0.25f * s;
}

// ---------------- sampling ----------------

static __device__ __forceinline__ float bilin(const float* fm, int h, int w,
                                              float x, float y, int c)
{
  float x0f = floorf(x), y0f = floorf(y);
  float wx = x - x0f, wy = y - y0f;
  int x0 = clampi((int)x0f, 0, w - 1);
  int x1 = x0 + 1 > w - 1 ? w - 1 : x0 + 1;
  int y0 = clampi((int)y0f, 0, h - 1);
  int y1 = y0 + 1 > h - 1 ? h - 1 : y0 + 1;
  float v00 = fm[((long)y0 * w + x0) * LATENT + c];
  float v01 = fm[((long)y0 * w + x1) * LATENT + c];
  float v10 = fm[((long)y1 * w + x0) * LATENT + c];
  float v11 = fm[((long)y1 * w + x1) * LATENT + c];
  return v00 * (1 - wx) * (1 - wy) + v01 * wx * (1 - wy) +
         v10 * (1 - wx) * wy + v11 * wx * wy;
}

__global__ __launch_bounds__(256)
void trackfeat_kernel(const float* __restrict__ fm, const float* __restrict__ queries,
                      float* __restrict__ tf)
{
  int idx = blockIdx.x * blockDim.x + threadIdx.x;
  if (idx >= NTRK * LATENT) return;
  int c = idx & 127, n = idx >> 7;
  int qf = (int)queries[n * 3 + 0];
  float qx = queries[n * 3 + 1] * 0.25f;
  float qy = queries[n * 3 + 2] * 0.25f;
  const float* fmt = fm + (long)qf * Hf * Wf * LATENT;
  tf[idx] = bilin(fmt, Hf, Wf, qx, qy, c);
}

__global__ __launch_bounds__(256)
void init_state_kernel(const float* __restrict__ queries, float* __restrict__ coords,
                       float* __restrict__ vis, float* __restrict__ conf)
{
  int idx = blockIdx.x * blockDim.x + threadIdx.x;   // t*NTRK + n
  if (idx >= T * NTRK) return;
  int n = idx % NTRK;
  coords[idx * 2 + 0] = queries[n * 3 + 1];
  coords[idx * 2 + 1] = queries[n * 3 + 2];
  vis[idx] = 0.5f;
  conf[idx] = 0.5f;
}

// one wave per (n,t,lvl,p) correlation sample, 4 channels per lane
__global__ __launch_bounds__(256)
void corr_kernel(const float* __restrict__ fm, const float* __restrict__ p1,
                 const float* __restrict__ p2, const float* __restrict__ p3,
                 const float* __restrict__ coords, const float* __restrict__ tf,
                 float* __restrict__ corr_raw)
{
  int wid = (blockIdx.x << 3) + (threadIdx.x >> 5);
  int lane = threadIdx.x & 31;
  const int TOTAL = NTRK * T * 4 * P;
  if (wid >= TOTAL) return;
  int p = wid % P;
  int lvl = (wid / P) & 3;
  int t = (wid / (P * 4)) % T;
  int n = wid / (P * 4 * T);

  float cx = coords[(t * NTRK + n) * 2 + 0] * 0.25f;
  float cy = coords[(t * NTRK + n) * 2 + 1] * 0.25f;
  float inv = 1.f / (float)(1 << lvl);
  float x = cx * inv + (float)(p % 7 - 3);
  float y = cy * inv + (float)(p / 7 - 3);
  const float* fmap; int h, w;
  if (lvl == 0)      { fmap = fm; h = 96; w = 128; }
  else if (lvl == 1) { fmap = p1; h = 48; w = 64;  }
  else if (lvl == 2) { fmap = p2; h = 24; w = 32;  }
  else               { fmap = p3; h = 12; w = 16;  }
  fmap += (long)t * h * w * LATENT;

  float x0f = floorf(x), y0f = floorf(y);
  float wx = x - x0f, wy = y - y0f;
  int x0 = clampi((int)x0f, 0, w - 1);
  int x1 = x0 + 1 > w - 1 ? w - 1 : x0 + 1;
  int y0 = clampi((int)y0f, 0, h - 1);
  int y1 = y0 + 1 > h - 1 ? h - 1 : y0 + 1;
  float w00 = (1 - wx) * (1 - wy), w01 = wx * (1 - wy);
  float w10 = (1 - wx) * wy,       w11 = wx * wy;
  const float* r0 = fmap + (long)y0 * w * LATENT;
  const float* r1 = fmap + (long)y1 * w * LATENT;
  const float* tfn = tf + n * LATENT;

  float acc = 0.f;
#pragma unroll
  for (int i = 0; i < 4; ++i) {
    int c = lane + 32 * i;
    float s = r0[x0 * LATENT + c] * w00 + r0[x1 * LATENT + c] * w01 +
              r1[x0 * LATENT + c] * w10 + r1[x1 * LATENT + c] * w11;
    acc += s * tfn[c];
  }
#pragma unroll
  for (int off = 16; off; off >>= 1) acc += __shfl_xor(acc, off, 32);
  if (lane == 0)
    corr_raw[(long)(n * T + t) * CORR_DIM + lvl * P + p] = acc * 0.08838834764831845f;
}

// ---------------- transformer pieces ----------------

// one wave per token: f32 layernorm -> f16 output (D = 384)
__global__ __launch_bounds__(256)
void ln_kernel(const float* __restrict__ x, const float* __restrict__ g,
               const float* __restrict__ b, half_t* __restrict__ out, int tokens)
{
  int wave = threadIdx.x >> 5, lane = threadIdx.x & 31;
  int tok = blockIdx.x * 8 + wave;
  if (tok >= tokens) return;
  const float* xr = x + (long)tok * HID;
  float vals[12];
  float s = 0.f;
#pragma unroll
  for (int i = 0; i < 12; ++i) { vals[i] = xr[lane + 32 * i]; s += vals[i]; }
#pragma unroll
  for (int off = 16; off; off >>= 1) s += __shfl_xor(s, off, 32);
  float mean = s * (1.f / HID);
  float q = 0.f;
#pragma unroll
  for (int i = 0; i < 12; ++i) { float d = vals[i] - mean; q += d * d; }
#pragma unroll
  for (int off = 16; off; off >>= 1) q += __shfl_xor(q, off, 32);
  float rstd = rsqrtf(q * (1.f / HID) + 1e-5f);
  half_t* orow = out + (long)tok * HID;
#pragma unroll
  for (int i = 0; i < 12; ++i) {
    int c = lane + 32 * i;
    orow[c] = (half_t)((vals[i] - mean) * rstd * g[c] + b[c]);
  }
}

// fused softmax attention: one block per (seq, head); K/V staged in LDS as f16
__global__ __launch_bounds__(64)
void attn_kernel(const float* __restrict__ qkv, float* __restrict__ o, int S)
{
  __shared__ half_t kls[SEQSP * HDIM];
  __shared__ half_t vls[SEQSP * HDIM];
  int seq = blockIdx.x >> 3;
  int h = blockIdx.x & 7;
  long base = (long)seq * S;
  for (int idx = threadIdx.x; idx < S * HDIM; idx += 64) {
    int s = idx / HDIM, d = idx - s * HDIM;
    const float* tp = qkv + (base + s) * 1152 + h * HDIM + d;
    kls[idx] = (half_t)tp[384];
    vls[idx] = (half_t)tp[768];
  }
  __syncthreads();
  const float scale = 0.14433756729740643f;  // 1/sqrt(48)
  for (int qi = threadIdx.x; qi < S; qi += 64) {
    float qreg[HDIM];
    const float* qp = qkv + (base + qi) * 1152 + h * HDIM;
#pragma unroll
    for (int d = 0; d < HDIM; ++d) qreg[d] = qp[d] * scale;
    float m = -1e30f;
    for (int s = 0; s < S; ++s) {
      float sc = 0.f;
      const half_t* kp = kls + s * HDIM;
#pragma unroll
      for (int d = 0; d < HDIM; ++d) sc += qreg[d] * (float)kp[d];
      m = fmaxf(m, sc);
    }
    float den = 0.f;
    float accv[HDIM];
#pragma unroll
    for (int d = 0; d < HDIM; ++d) accv[d] = 0.f;
    for (int s = 0; s < S; ++s) {
      float sc = 0.f;
      const half_t* kp = kls + s * HDIM;
#pragma unroll
      for (int d = 0; d < HDIM; ++d) sc += qreg[d] * (float)kp[d];
      float e = expf(sc - m);
      den += e;
      const half_t* vp = vls + s * HDIM;
#pragma unroll
      for (int d = 0; d < HDIM; ++d) accv[d] += e * (float)vp[d];
    }
    float invd = 1.f / den;
    float* op = o + (base + qi) * HID + h * HDIM;
#pragma unroll
    for (int d = 0; d < HDIM; ++d) op[d] = accv[d] * invd;
  }
}

// ---------------- glue kernels ----------------

// corrO leading dim 224 (padded)
__global__ __launch_bounds__(256)
void tin_kernel(const float* __restrict__ corrO, const float* __restrict__ coords,
                const float* __restrict__ queries, const float* __restrict__ vis,
                const float* __restrict__ conf, half_t* __restrict__ tin)
{
  int row = blockIdx.x * blockDim.x + threadIdx.x;  // n*T + t
  if (row >= TOKENS) return;
  int t = row % T, n = row / T;
  half_t* dst = tin + (long)row * 256;
  const float* cr = corrO + (long)row * 224;
  for (int c = 0; c < CORR_DIM; ++c) dst[c] = (half_t)cr[c];
  float dx = coords[(t * NTRK + n) * 2 + 0] - queries[n * 3 + 1];
  float dy = coords[(t * NTRK + n) * 2 + 1] - queries[n * 3 + 2];
  float f = 1.f;
  for (int i = 0; i < 11; ++i) {
    dst[196 + i] = (half_t)sinf(dx * f);
    dst[207 + i] = (half_t)cosf(dx * f);
    dst[218 + i] = (half_t)sinf(dy * f);
    dst[229 + i] = (half_t)cosf(dy * f);
    f *= 2.f;
  }
  dst[240] = (half_t)vis[t * NTRK + n];
  dst[241] = (half_t)conf[t * NTRK + n];
  for (int c = 242; c < 256; ++c) dst[c] = (half_t)0.f;
}

__global__ __launch_bounds__(256)
void build_xsp_kernel(const float* __restrict__ tok, const float* __restrict__ virt,
                      float* __restrict__ xsp)
{
  long idx = (long)blockIdx.x * blockDim.x + threadIdx.x;
  const long total = (long)TOKSP * HID;
  if (idx >= total) return;
  int c = (int)(idx % HID);
  long r = idx / HID;
  int s = (int)(r % SEQSP);
  int t = (int)(r / SEQSP);
  xsp[idx] = (s < NTRK) ? tok[((long)s * T + t) * HID + c]
                        : virt[(long)(s - NTRK) * HID + c];
}

__global__ __launch_bounds__(256)
void scatter_xsp_kernel(const float* __restrict__ xsp, float* __restrict__ tok)
{
  long idx = (long)blockIdx.x * blockDim.x + threadIdx.x;
  const long total = (long)TOKENS * HID;
  if (idx >= total) return;
  int c = (int)(idx % HID);
  long r = idx / HID;
  int t = (int)(r % T);
  int n = (int)(r / T);
  tok[((long)n * T + t) * HID + c] = xsp[((long)t * SEQSP + n) * HID + c];
}

// delta leading dim 32 (padded from 4)
__global__ __launch_bounds__(256)
void update_kernel(const float* __restrict__ delta, float* __restrict__ coords,
                   float* __restrict__ vis, float* __restrict__ conf)
{
  int idx = blockIdx.x * blockDim.x + threadIdx.x;  // t*NTRK + n
  if (idx >= T * NTRK) return;
  int t = idx / NTRK, n = idx % NTRK;
  const float* d = delta + ((long)n * T + t) * 32;
  coords[idx * 2 + 0] += d[0];
  coords[idx * 2 + 1] += d[1];
  vis[idx]  = 1.f / (1.f + expf(-(vis[idx]  + d[2])));
  conf[idx] = 1.f / (1.f + expf(-(conf[idx] + d[3])));
}

__global__ __launch_bounds__(256)
void finalize_kernel(const float* __restrict__ coords, const float* __restrict__ vis,
                     const float* __restrict__ conf, float* __restrict__ out)
{
  int idx = blockIdx.x * blockDim.x + threadIdx.x;
  if (idx >= T * NTRK) return;
  out[idx * 4 + 0] = coords[idx * 2 + 0];
  out[idx * 4 + 1] = coords[idx * 2 + 1];
  out[idx * 4 + 2] = vis[idx];
  out[idx * 4 + 3] = conf[idx];
}

// ---------------- host driver ----------------

static inline long cdiv(long a, long b) { return (a + b - 1) / b; }

extern "C" void kernel_launch(void* const* d_in, const int* in_sizes, int n_in,
                              void* d_out, int out_size, void* d_ws, size_t ws_size,
                              hipStream_t stream)
{
  (void)in_sizes; (void)n_in; (void)out_size; (void)ws_size;
  const float* video   = (const float*)d_in[0];
  const float* queries = (const float*)d_in[1];
  const float* f_w1 = (const float*)d_in[2];
  const float* f_b1 = (const float*)d_in[3];
  const float* f_w2 = (const float*)d_in[4];
  const float* f_b2 = (const float*)d_in[5];
  const float* f_w3 = (const float*)d_in[6];
  const float* f_b3 = (const float*)d_in[7];
  const float* c_w1 = (const float*)d_in[8];
  const float* c_b1 = (const float*)d_in[9];
  const float* c_w2 = (const float*)d_in[10];
  const float* c_b2 = (const float*)d_in[11];
  const float* uf_inw  = (const float*)d_in[12];
  const float* uf_inb  = (const float*)d_in[13];
  const float* uf_outw = (const float*)d_in[14];
  const float* uf_outb = (const float*)d_in[15];
  const float* uf_virt = (const float*)d_in[16];
  auto blk = [&](int bi, int j) -> const float* {
    return (const float*)d_in[17 + bi * 12 + j];
  };
  // block j: 0 ln1_s 1 ln1_b 2 qkv_w 3 qkv_b 4 proj_w 5 proj_b
  //          6 ln2_s 7 ln2_b 8 mlp_w1 9 mlp_b1 10 mlp_w2 11 mlp_b2

  char* ws = (char*)d_ws;
  size_t off = 0;
  auto alloc = [&](size_t bytes) -> char* {
    off = (off + 255) & ~(size_t)255;
    char* p = ws + off;
    off += bytes;
    return p;
  };

  half_t* c1buf = (half_t*)alloc((size_t)T * H1 * W1 * C1CH * 2);   // 100.7 MB
  half_t* c2buf = (half_t*)alloc((size_t)T * Hf * Wf * 128 * 2);    // 50.3 MB
  float* fm   = (float*)c1buf;                                      // alias (c1 dead after conv2)
  float* pyr1 = (float*)c2buf;                                      // alias (c2 dead after conv3)
  float* pyr2 = pyr1 + (size_t)T * 48 * 64 * 128;
  float* pyr3 = pyr2 + (size_t)T * 24 * 32 * 128;

  half_t* whW3  = (half_t*)alloc((size_t)128 * 128 * 2);
  half_t* whC1  = (half_t*)alloc((size_t)224 * 416 * 2);
  half_t* whC2  = (half_t*)alloc((size_t)416 * 224 * 2);
  half_t* whIn  = (half_t*)alloc((size_t)256 * 384 * 2);
  half_t* whOut = (half_t*)alloc((size_t)384 * 32 * 2);
  half_t* whB[6][4];
  for (int b = 0; b < 6; ++b) {
    whB[b][0] = (half_t*)alloc((size_t)384 * 1152 * 2);
    whB[b][1] = (half_t*)alloc((size_t)384 * 384 * 2);
    whB[b][2] = (half_t*)alloc((size_t)384 * 1536 * 2);
    whB[b][3] = (half_t*)alloc((size_t)1536 * 384 * 2);
  }
  float* tf      = (float*)alloc((size_t)NTRK * 128 * 4);
  float* coords  = (float*)alloc((size_t)T * NTRK * 2 * 4);
  float* vis     = (float*)alloc((size_t)T * NTRK * 4);
  float* conf    = (float*)alloc((size_t)T * NTRK * 4);
  float* tok     = (float*)alloc((size_t)TOKENS * HID * 4);
  float* xsp     = (float*)alloc((size_t)TOKSP * HID * 4);
  float* qkvF    = (float*)alloc((size_t)TOKSP * 1152 * 4);
  float* oF      = (float*)alloc((size_t)TOKSP * HID * 4);
  half_t* Ahalf  = (half_t*)alloc((size_t)TOKSP * 1536 * 2);
  float* tmpF    = (float*)alloc((size_t)TOKSP * 1536 * 4);
  float* corrRaw = (float*)alloc((size_t)TOKENS * CORR_DIM * 4);
  float* corrO   = (float*)alloc((size_t)TOKENS * 224 * 4);
  half_t* tinH   = (half_t*)alloc((size_t)TOKENS * 256 * 2);

  auto castW = [&](const float* w, half_t* dst, int K, int Nc, int Kp, int Np) {
    long total = (long)Kp * Np;
    castW_kernel<<<(unsigned)cdiv(total, 256), 256, 0, stream>>>(w, dst, K, Nc, Kp, Np);
  };
  auto castA = [&](const float* src, half_t* dst, long rows, int cols, int srcld, int pcols) {
    long total = rows * pcols;
    cast_pad_kernel<<<(unsigned)cdiv(total, 256), 256, 0, stream>>>(src, dst, rows, cols, srcld, pcols);
  };
  auto gemm = [&](const half_t* A, const half_t* Bm, const float* bias, float* C,
                  long M, int Nc, int Np, int Kp, int act, int resid) {
    long tiles = (M / 16) * (Np / 32);
    gemm_wmma_kernel<<<(unsigned)cdiv(tiles, 8), 256, 0, stream>>>(
        A, Bm, bias, C, (int)M, Nc, Np, Kp, act, resid);
  };

  // ---- weight conversion (f32 -> K/N padded f16, row-major Kp x Np) ----
  castW(f_w3, whW3, 128, 128, 128, 128);
  castW(c_w1, whC1, 196, 392, 224, 416);
  castW(c_w2, whC2, 392, 196, 416, 224);
  castW(uf_inw, whIn, 242, 384, 256, 384);
  castW(uf_outw, whOut, 384, 4, 384, 32);
  for (int b = 0; b < 6; ++b) {
    castW(blk(b, 2),  whB[b][0], 384, 1152, 384, 1152);
    castW(blk(b, 4),  whB[b][1], 384, 384, 384, 384);
    castW(blk(b, 8),  whB[b][2], 384, 1536, 384, 1536);
    castW(blk(b, 10), whB[b][3], 1536, 384, 1536, 384);
  }

  // ---- feature net ----
  {
    long t1 = (long)T * H1 * W1 * C1CH;
    conv1_kernel<<<(unsigned)cdiv(t1, 256), 256, 0, stream>>>(video, f_w1, f_b1, c1buf);
    long t2 = (long)T * Hf * Wf * 128;
    conv2_kernel<<<(unsigned)cdiv(t2, 256), 256, 0, stream>>>(c1buf, f_w2, f_b2, c2buf);
    gemm(c2buf, whW3, f_b3, fm, (long)T * Hf * Wf, 128, 128, 128, 0, 0);  // 1x1 conv via WMMA
    long p1t = (long)T * 48 * 64 * 128;
    avgpool_kernel<<<(unsigned)cdiv(p1t, 256), 256, 0, stream>>>(fm, pyr1, 96, 128, p1t);
    long p2t = (long)T * 24 * 32 * 128;
    avgpool_kernel<<<(unsigned)cdiv(p2t, 256), 256, 0, stream>>>(pyr1, pyr2, 48, 64, p2t);
    long p3t = (long)T * 12 * 16 * 128;
    avgpool_kernel<<<(unsigned)cdiv(p3t, 256), 256, 0, stream>>>(pyr2, pyr3, 24, 32, p3t);
    trackfeat_kernel<<<(unsigned)cdiv(NTRK * 128, 256), 256, 0, stream>>>(fm, queries, tf);
    init_state_kernel<<<(unsigned)cdiv(T * NTRK, 256), 256, 0, stream>>>(queries, coords, vis, conf);
  }

  auto attn_block = [&](float* x, long Stot, int S, int nSeq, int b) {
    unsigned lnb = (unsigned)cdiv(Stot, 8);
    ln_kernel<<<lnb, 256, 0, stream>>>(x, blk(b, 0), blk(b, 1), Ahalf, (int)Stot);
    gemm(Ahalf, whB[b][0], blk(b, 3), qkvF, Stot, 1152, 1152, 384, 0, 0);
    attn_kernel<<<(unsigned)(nSeq * HEADS), 64, 0, stream>>>(qkvF, oF, S);
    castA(oF, Ahalf, Stot, HID, HID, HID);
    gemm(Ahalf, whB[b][1], blk(b, 5), x, Stot, 384, 384, 384, 0, 1);      // + residual
    ln_kernel<<<lnb, 256, 0, stream>>>(x, blk(b, 6), blk(b, 7), Ahalf, (int)Stot);
    gemm(Ahalf, whB[b][2], blk(b, 9), tmpF, Stot, 1536, 1536, 384, 1, 0); // gelu
    castA(tmpF, Ahalf, Stot, 1536, 1536, 1536);
    gemm(Ahalf, whB[b][3], blk(b, 11), x, Stot, 384, 384, 1536, 0, 1);    // + residual
  };

  // ---- iterative refinement ----
  for (int it = 0; it < ITERS; ++it) {
    const long CW = (long)NTRK * T * 4 * P;  // 401408 waves
    corr_kernel<<<(unsigned)cdiv(CW, 8), 256, 0, stream>>>(fm, pyr1, pyr2, pyr3,
                                                           coords, tf, corrRaw);
    castA(corrRaw, Ahalf, TOKENS, 196, 196, 224);
    gemm(Ahalf, whC1, c_b1, tmpF, TOKENS, 392, 416, 224, 1, 0);           // gelu
    castA(tmpF, Ahalf, TOKENS, 392, 416, 416);
    gemm(Ahalf, whC2, c_b2, corrO, TOKENS, 196, 224, 416, 0, 0);
    tin_kernel<<<(unsigned)cdiv(TOKENS, 256), 256, 0, stream>>>(corrO, coords, queries,
                                                                vis, conf, tinH);
    gemm(tinH, whIn, uf_inb, tok, TOKENS, 384, 384, 256, 0, 0);

    for (int b = 0; b < 3; ++b)
      attn_block(tok, TOKENS, T, NTRK, b);                           // time attention

    for (int b = 3; b < 6; ++b) {                                    // space attention
      long bt = (long)TOKSP * HID;
      build_xsp_kernel<<<(unsigned)cdiv(bt, 256), 256, 0, stream>>>(tok, uf_virt, xsp);
      attn_block(xsp, TOKSP, SEQSP, T, b);
      long st = (long)TOKENS * HID;
      scatter_xsp_kernel<<<(unsigned)cdiv(st, 256), 256, 0, stream>>>(xsp, tok);
    }

    castA(tok, Ahalf, TOKENS, 384, 384, 384);
    gemm(Ahalf, whOut, uf_outb, tmpF, TOKENS, 4, 32, 384, 0, 0);     // delta (ld 32)
    update_kernel<<<(unsigned)cdiv(T * NTRK, 256), 256, 0, stream>>>(tmpF, coords, vis, conf);
  }

  finalize_kernel<<<(unsigned)cdiv(T * NTRK, 256), 256, 0, stream>>>(coords, vis, conf,
                                                                     (float*)d_out);
}
